// Cross_Attention_61830349193402
// MI455X (gfx1250) — matrix operand order repaired and checked by reading, not compiled, not run on hardware
//
#include <hip/hip_runtime.h>

#define NPOS   16384
#define CCH    256
#define HEADS  8
#define DK     32
#define BATCH  8

typedef __attribute__((ext_vector_type(16))) __bf16 v16bf;
typedef __attribute__((ext_vector_type(8)))  __bf16 v8bf;
typedef __attribute__((ext_vector_type(8)))  float  v8f;

// Issue 4x 16B async global->LDS copies for one 64B Weff row slice.
// ASYNCcnt tracked; consumed via s_wait_asynccnt + workgroup barrier.
__device__ __forceinline__ void async_copy_row64(const void* gsrc, unsigned lds_off) {
#pragma unroll
  for (int i = 0; i < 4; ++i) {
    unsigned long long ga = (unsigned long long)gsrc + (unsigned)(i * 16);
    unsigned la = lds_off + (unsigned)(i * 16);
    asm volatile("global_load_async_to_lds_b128 %0, %1, off"
                 :: "v"(la), "v"(ga) : "memory");
  }
}

// ---------------------------------------------------------------------------
// Kernel A1: per-(b,h,d) row max & 1/sum(exp) over the N=16384 spatial axis.
// ---------------------------------------------------------------------------
__global__ __launch_bounds__(256) void krowstats(const float* __restrict__ x2,
                                                 float* __restrict__ rowmax,
                                                 float* __restrict__ rowinv) {
  int bh = blockIdx.x;
  int b = bh >> 3, h = bh & 7;
  int d = threadIdx.x & 31;
  int part = threadIdx.x >> 5;
  const float* base = x2 + ((size_t)b * NPOS) * CCH + h * DK + d;
  float m = -3.0e38f, s = 0.0f;
  for (int n = part; n < NPOS; n += 8) {
    float x = base[(size_t)n * CCH];
    float mn = fmaxf(m, x);
    s = s * __expf(m - mn) + __expf(x - mn);
    m = mn;
  }
  __shared__ float sm[8][32], ss[8][32];
  sm[part][d] = m;
  ss[part][d] = s;
  __syncthreads();
  if (threadIdx.x < 32) {
    float M = sm[0][d], S = ss[0][d];
#pragma unroll
    for (int p = 1; p < 8; ++p) {
      float m2 = sm[p][d], s2 = ss[p][d];
      float mn = fmaxf(M, m2);
      S = S * __expf(M - mn) + s2 * __expf(m2 - mn);
      M = mn;
    }
    rowmax[bh * 32 + d] = M;
    rowinv[bh * 32 + d] = 1.0f / S;
  }
}

// ---------------------------------------------------------------------------
// Kernel A2: ctx[b,h,d,v] = (1/S_d) * sum_n exp(x2[d,n]-max_d) * x2[v,n]
// ---------------------------------------------------------------------------
__global__ __launch_bounds__(256) void kctx(const float* __restrict__ x2,
                                            const float* __restrict__ rowmax,
                                            const float* __restrict__ rowinv,
                                            float* __restrict__ ctx) {
  int bh = blockIdx.x;
  int b = bh >> 3, h = bh & 7;
  int lane = threadIdx.x & 31;
  int wave = threadIdx.x >> 5;
  float mlane = rowmax[bh * 32 + lane];
  const float* base = x2 + ((size_t)b * NPOS) * CCH + h * DK + lane;
  float acc[32];
#pragma unroll
  for (int d = 0; d < 32; ++d) acc[d] = 0.0f;
  for (int n = wave; n < NPOS; n += 8) {
    float x = base[(size_t)n * CCH];
    float e = __expf(x - mlane);
#pragma unroll
    for (int d = 0; d < 32; ++d) acc[d] += __shfl(e, d, 32) * x;
  }
  __shared__ float partsum[8][32][32];
#pragma unroll
  for (int d = 0; d < 32; ++d) partsum[wave][d][lane] = acc[d];
  __syncthreads();
#pragma unroll
  for (int it = 0; it < 4; ++it) {
    int idx = it * 256 + threadIdx.x;
    int d = idx >> 5;
    float s = 0.0f;
#pragma unroll
    for (int w = 0; w < 8; ++w) s += partsum[w][d][idx & 31];
    ctx[(size_t)bh * 1024 + idx] = s * rowinv[bh * 32 + d];
  }
}

// ---------------------------------------------------------------------------
// Kernel B: Weff_b[o, h*32+d] = sum_v W[o, h*32+v]*ctx[d,v], emitted in bf16
// so the main kernel streams it straight into LDS with async copies.
// ---------------------------------------------------------------------------
__global__ __launch_bounds__(256) void kweff(const float* __restrict__ conv_w,
                                             const float* __restrict__ ctx,
                                             __bf16* __restrict__ weffbf) {
  int bh = blockIdx.x;
  int b = bh >> 3, h = bh & 7;
  __shared__ float cx[32][32];
#pragma unroll
  for (int it = 0; it < 4; ++it) {
    int idx = it * 256 + threadIdx.x;
    ((float*)cx)[idx] = ctx[(size_t)bh * 1024 + idx];
  }
  __syncthreads();
  int o = threadIdx.x;
  float acc[32];
#pragma unroll
  for (int d = 0; d < 32; ++d) acc[d] = 0.0f;
  const float* wrow = conv_w + (size_t)o * CCH + h * DK;
#pragma unroll 4
  for (int v = 0; v < 32; ++v) {
    float wv = wrow[v];
#pragma unroll
    for (int d = 0; d < 32; ++d) acc[d] += wv * cx[d][v];
  }
  __bf16* dst = weffbf + ((size_t)b * CCH + o) * CCH + h * DK;
#pragma unroll
  for (int d = 0; d < 32; ++d) dst[d] = (__bf16)acc[d];
}

// ---------------------------------------------------------------------------
// Kernel C: per WG = 32 positions of one batch.
//  phase1: head-wise channel softmax of x1 -> LDS bf16 A tile (32 x 256)
//  phase2: WMMA GEMM vs Weff^T; Weff k-slices double-buffered in LDS via
//          global_load_async_to_lds_b128 (ASYNCcnt) overlapping the MMAs
//  phase3: bias + LayerNorm fully in registers (shfl_xor + 512B LDS exchange)
// ---------------------------------------------------------------------------
__global__ __launch_bounds__(256) void kmain(const float* __restrict__ x1,
                                             const __bf16* __restrict__ weffbf,
                                             const float* __restrict__ conv_b,
                                             const float* __restrict__ ln_g,
                                             const float* __restrict__ ln_b,
                                             float* __restrict__ out) {
  __shared__ __align__(64) unsigned char smem[48 * 1024];
  __bf16 (*Aq)[CCH] = (__bf16(*)[CCH])smem;                 // [32][256] 16KB
  __bf16* Wb0 = (__bf16*)(smem + 16 * 1024);                // [256][32] 16KB
  __bf16* Wb1 = (__bf16*)(smem + 32 * 1024);                // [256][32] 16KB
  __shared__ float lnS[32][4], lnQ[32][4];                  // LN cross-wave

  int b  = blockIdx.y;
  int n0 = blockIdx.x * 32;
  int tid = threadIdx.x;

  // Kick off async staging of Weff k-slice 0 (bf16, 64B per row per thread).
  const __bf16* wrow = weffbf + ((size_t)b * CCH + tid) * CCH;
  unsigned lds0 = (unsigned)(uintptr_t)(const void*)(Wb0 + tid * DK);
  unsigned lds1 = (unsigned)(uintptr_t)(const void*)(Wb1 + tid * DK);
  async_copy_row64(wrow, lds0);

  // ---- phase 1: softmax over each head's 32 channels, store bf16 A ----
  {
    int pos = tid >> 3, hh = tid & 7;
    const float* src = x1 + ((size_t)b * NPOS + n0 + pos) * CCH + hh * DK;
    float v[32];
#pragma unroll
    for (int i = 0; i < 8; ++i) {
      float4 f = ((const float4*)src)[i];
      v[4 * i] = f.x; v[4 * i + 1] = f.y; v[4 * i + 2] = f.z; v[4 * i + 3] = f.w;
    }
    float m = v[0];
#pragma unroll
    for (int i = 1; i < 32; ++i) m = fmaxf(m, v[i]);
    float s = 0.0f;
#pragma unroll
    for (int i = 0; i < 32; ++i) { v[i] = __expf(v[i] - m); s += v[i]; }
    float inv = 1.0f / s;
#pragma unroll
    for (int i = 0; i < 32; ++i) Aq[pos][hh * DK + i] = (__bf16)(v[i] * inv);
  }
  asm volatile("s_wait_asynccnt 0x0" ::: "memory");  // slice 0 resident
  __syncthreads();                                   // + Aq visible

  // ---- phase 2: WMMA GEMM D[32 pos][256 ch] ----
  int lane  = tid & 31;
  int wave  = tid >> 5;
  int mtile = wave >> 2;                 // 0..1 : 16-position tile
  int oq    = wave & 3;                  // 64-output-channel quarter
  int klo_a = (lane < 16) ? 0 : 16;      // contiguous k = 16h + s (matches B)
  int klo_b = (lane < 16) ? 0 : 16;      // contiguous k = 16h + s (matches A)
  int col   = lane & 15;

  v8f acc[4];
#pragma unroll
  for (int t = 0; t < 4; ++t) {
    v8f z = {0.f, 0.f, 0.f, 0.f, 0.f, 0.f, 0.f, 0.f};
    acc[t] = z;
  }

  for (int kk = 0; kk < 8; ++kk) {
    const __bf16* WbCur = (kk & 1) ? Wb1 : Wb0;
    if (kk < 7) {
      // prefetch slice kk+1 into the other buffer (freed by last barrier)
      async_copy_row64(wrow + (kk + 1) * DK, (kk & 1) ? lds0 : lds1);
      asm volatile("s_wait_asynccnt 0x4" ::: "memory");  // slice kk done
    } else {
      asm volatile("s_wait_asynccnt 0x0" ::: "memory");
    }
    __syncthreads();

    const __bf16* ap = &Aq[mtile * 16 + col][kk * DK];
    v8bf alo = *(const v8bf*)(ap + klo_a);
    v8bf ahi = *(const v8bf*)(ap + klo_a + 8);
    v16bf afrag = __builtin_shufflevector(alo, ahi,
        0, 1, 2, 3, 4, 5, 6, 7, 8, 9, 10, 11, 12, 13, 14, 15);

#pragma unroll
    for (int t = 0; t < 4; ++t) {
      int o = oq * 64 + t * 16 + col;
      v16bf bfrag = *(const v16bf*)&WbCur[o * DK + klo_b];
      acc[t] = __builtin_amdgcn_wmma_f32_16x16x32_bf16(
          false, afrag, false, bfrag, (short)0, acc[t], false, false);
    }
    __syncthreads();                     // all waves done with WbCur
  }

  // ---- phase 3: bias + LayerNorm in registers ----
  int rbase = (lane < 16) ? 0 : 8;       // D layout: row = vgpr + rbase
  float s8[8], q8[8];
#pragma unroll
  for (int v = 0; v < 8; ++v) { s8[v] = 0.0f; q8[v] = 0.0f; }
#pragma unroll
  for (int t = 0; t < 4; ++t) {
    int o = oq * 64 + t * 16 + col;
    float bias = conv_b[o];
#pragma unroll
    for (int v = 0; v < 8; ++v) {
      float val = acc[t][v] + bias;
      acc[t][v] = val;
      s8[v] += val;
      q8[v] += val * val;
    }
  }
#pragma unroll
  for (int v = 0; v < 8; ++v) {
#pragma unroll
    for (int msk = 1; msk < 16; msk <<= 1) {
      s8[v] += __shfl_xor(s8[v], msk, 16);
      q8[v] += __shfl_xor(q8[v], msk, 16);
    }
  }
  if (col == 0) {                        // lane 0 & lane 16 of each wave
#pragma unroll
    for (int v = 0; v < 8; ++v) {
      lnS[mtile * 16 + rbase + v][oq] = s8[v];
      lnQ[mtile * 16 + rbase + v][oq] = q8[v];
    }
  }
  __syncthreads();

  float mu8[8], rs8[8];
#pragma unroll
  for (int v = 0; v < 8; ++v) {
    int row = mtile * 16 + rbase + v;
    float S = lnS[row][0] + lnS[row][1] + lnS[row][2] + lnS[row][3];
    float Q = lnQ[row][0] + lnQ[row][1] + lnQ[row][2] + lnQ[row][3];
    float mu = S * (1.0f / 256.0f);
    mu8[v] = mu;
    rs8[v] = rsqrtf(Q * (1.0f / 256.0f) - mu * mu + 1e-5f);
  }
#pragma unroll
  for (int t = 0; t < 4; ++t) {
    int o = oq * 64 + t * 16 + col;
    float g = ln_g[o], bb = ln_b[o];
#pragma unroll
    for (int v = 0; v < 8; ++v) {
      int row = mtile * 16 + rbase + v;
      out[((size_t)b * NPOS + n0 + row) * CCH + o] =
          (acc[t][v] - mu8[v]) * rs8[v] * g + bb;
    }
  }
}

// ---------------------------------------------------------------------------
extern "C" void kernel_launch(void* const* d_in, const int* in_sizes, int n_in,
                              void* d_out, int out_size, void* d_ws, size_t ws_size,
                              hipStream_t stream) {
  const float* x1     = (const float*)d_in[0];
  const float* x2     = (const float*)d_in[1];
  const float* conv_w = (const float*)d_in[2];
  const float* conv_b = (const float*)d_in[3];
  const float* ln_g   = (const float*)d_in[4];
  const float* ln_b   = (const float*)d_in[5];
  float* out = (float*)d_out;

  float*  ws     = (float*)d_ws;
  float*  rowmax = ws;                         //  2048 f32
  float*  rowinv = ws + 2048;                  //  2048 f32
  float*  ctx    = ws + 4096;                  // 64*1024 f32
  __bf16* weffbf = (__bf16*)(ws + 4096 + 64 * 1024);  // 8*256*256 bf16 (1MB)

  krowstats<<<dim3(BATCH * HEADS), dim3(256), 0, stream>>>(x2, rowmax, rowinv);
  kctx<<<dim3(BATCH * HEADS), dim3(256), 0, stream>>>(x2, rowmax, rowinv, ctx);
  kweff<<<dim3(BATCH * HEADS), dim3(256), 0, stream>>>(conv_w, ctx, weffbf);
  kmain<<<dim3(NPOS / 32, BATCH), dim3(256), 0, stream>>>(x1, weffbf, conv_b,
                                                          ln_g, ln_b, out);
}
